// SelfAttention_74139725463800
// MI455X (gfx1250) — matrix operand order repaired and checked by reading, not compiled, not run on hardware
//
#include <hip/hip_runtime.h>
#include <math.h>

// ---------------------------------------------------------------------------
// CDNA5 (gfx1250) WMMA self-attention pipeline, TDM-fed.
//
// ~190 GFLOP vs ~130MB traffic @ 23.3 TB/s => strongly compute bound; all four
// GEMMs run on v_wmma_f32_16x16x32_f16 (wave32). Pure-f16 panels are staged
// LDS-direct by the Tensor Data Mover (tensor_load_to_lds, TENSORcnt), double
// buffered so DMA of chunk i+1 overlaps WMMA on chunk i. f32 sources are
// register-staged in two phases (all loads, then all cvt+stores) so the
// global_load_b128s pipeline under a single wait.
//
// Workspace (f16): Qh[B][N][D] 32MB | Qth[B][D][N] 32MB | Mixh[B][N][D] 32MB
//                  S/P[B][N][N] 128MB (scores overwritten by weights in place)
// ---------------------------------------------------------------------------

typedef __attribute__((ext_vector_type(16))) _Float16 v16h;
typedef __attribute__((ext_vector_type(8)))  _Float16 v8h;
typedef __attribute__((ext_vector_type(4)))  _Float16 v4h;
typedef __attribute__((ext_vector_type(8)))  float    v8f;
typedef unsigned int u32x4 __attribute__((ext_vector_type(4)));
typedef int          i32x4 __attribute__((ext_vector_type(4)));
typedef int          i32x8 __attribute__((ext_vector_type(8)));

#define KC       64     // K-chunk (two WMMA k-steps)
#define BM       64     // block tile rows (A panel rows)
#define BN       128    // block tile cols (B panel rows)
#define NTHREADS 256    // 8 wave32 waves: 2 (M) x 4 (N), each wave 32x32 out

#if __has_include(<hip/amd_detail/amd_gfx1250_TDM.h>)
#define CDNA5_TDM_6ARG 1
#else
#define CDNA5_TDM_6ARG 0
#endif

union HFrag { v16h v; v8h h[2]; };

// Low 32 bits of a flat pointer into LDS == LDS byte address (ISA 10.2).
__device__ __forceinline__ unsigned lds_off(const void* p) {
  return (unsigned)(uintptr_t)p;
}

// ---------------------------------------------------------------------------
// Tensor Data Mover: DMA a rows x KC f16 tile (row-major, row stride ld_elems
// halves) from global memory into LDS at lds_addr. D# per ISA ch.8:
//  g0: count=1 | lds_addr<<32 | global_addr@64 | type=2@126
//  g1: data_size=1(2B)@16, tensor_dim0@48, tensor_dim1@80 (huge: no OOB clip),
//      tile_dim0=KC@112, tile_dim1=rows@128, tensor_dim0_stride@160
// Issued by one wave (EXEC ignored); completion via s_wait_tensorcnt.
// ---------------------------------------------------------------------------
__device__ __forceinline__ void tdm_load_tile(unsigned lds_addr, const void* gsrc,
                                              unsigned ld_elems, unsigned rows) {
  union { u32x4 v; unsigned long long q[2]; } g0;
  union { i32x8 v; unsigned long long q[4]; } g1;
  const unsigned long long ga = (unsigned long long)(uintptr_t)gsrc;
  const unsigned long long TD = 1ull << 30;              // huge tensor dims
  g0.q[0] = 1ull | ((unsigned long long)lds_addr << 32); // count=1, lds_addr
  g0.q[1] = (ga & 0x01FFFFFFFFFFFFFFull) | (2ull << 62); // global_addr, type=2
  g1.q[0] = (1ull << 16)                                 // data_size = 2 bytes
          | ((TD & 0xFFFFull) << 48);                    // tensor_dim0[15:0]
  g1.q[1] = (TD >> 16)                                   // tensor_dim0[31:16]
          | (TD << 16)                                   // tensor_dim1 @bit80
          | ((unsigned long long)KC << 48);              // tile_dim0 @bit112
  g1.q[2] = (unsigned long long)rows                     // tile_dim1 @bit128
          | ((unsigned long long)ld_elems << 32);        // dim0_stride[31:0] @bit160
  g1.q[3] = 0ull;                                        // stride hi, dim1_stride
  const i32x4 z4 = {0, 0, 0, 0};
#if CDNA5_TDM_6ARG
  const i32x8 z8 = {0, 0, 0, 0, 0, 0, 0, 0};
  __builtin_amdgcn_tensor_load_to_lds(g0.v, g1.v, z4, z4, z8, 0);
#else
  __builtin_amdgcn_tensor_load_to_lds(g0.v, g1.v, z4, z4, 0);
#endif
}

// A fragment, 16x32 f16 (ISA 7.12.2): lanes 0-15 row=lane, K 0..7/16..23;
// lanes 16-31 row=lane-16, K 8..15/24..31. Panel row-major [row][KC].
__device__ __forceinline__ v16h load_a_frag(const _Float16* panel, int r0, int k0, int lane) {
  const int row  = r0 + (lane & 15);
  const int ksel = (lane >> 4) << 3;
  HFrag f;
  f.h[0] = *(const v8h*)(panel + row * KC + k0 + ksel);
  f.h[1] = *(const v8h*)(panel + row * KC + k0 + 16 + ksel);
  return f.v;
}

// B fragment, 32x16 f16: lanes 0-15 col=lane K 0..7/16..23; lanes 16-31 col-16 K 8..15/24..31.
// Panel staged K-transposed as [n][KC] -> two runs of 8 contiguous halves per lane, 16 apart.
__device__ __forceinline__ v16h load_b_frag(const _Float16* panelT, int n0, int k0, int lane) {
  const int col  = n0 + (lane & 15);
  const int koff = (lane >> 4) << 3;
  const _Float16* p = panelT + col * KC + k0 + koff;
  HFrag f;
  f.h[0] = *(const v8h*)(p);
  f.h[1] = *(const v8h*)(p + 16);
  return f.v;
}

__device__ __forceinline__ v8f wmma(v16h a, v16h b, v8f c) {
  return __builtin_amdgcn_wmma_f32_16x16x32_f16(false, a, false, b, (short)0, c, false, false);
}

// One K-chunk for a wave's 32x32 slice: 8x v_wmma, each frag reused twice.
__device__ __forceinline__ void mma_block(const _Float16* As, const _Float16* Bs,
                                          v8f acc[2][2], int wm, int wn, int lane) {
#pragma unroll
  for (int kk = 0; kk < KC; kk += 32) {
    const v16h a0 = load_a_frag(As, wm * 32,      kk, lane);
    const v16h a1 = load_a_frag(As, wm * 32 + 16, kk, lane);
    const v16h b0 = load_b_frag(Bs, wn * 32,      kk, lane);
    const v16h b1 = load_b_frag(Bs, wn * 32 + 16, kk, lane);
    acc[0][0] = wmma(a0, b0, acc[0][0]);
    acc[0][1] = wmma(a0, b1, acc[0][1]);
    acc[1][0] = wmma(a1, b0, acc[1][0]);
    acc[1][1] = wmma(a1, b1, acc[1][1]);
  }
}

// Stage ROWS x KC panel from f32 source, converting to f16. Two phases so the
// global_load_b128s all issue before a single loadcnt wait.
template <int ROWS>
__device__ __forceinline__ void stage_f32(_Float16* dst, const float* src, int ld, int tid) {
  constexpr int ITERS = (ROWS * KC / 4) / NTHREADS;
  float4 v[ITERS];
#pragma unroll
  for (int i = 0; i < ITERS; ++i) {
    const int c = tid + i * NTHREADS;
    v[i] = *(const float4*)(src + (size_t)(c >> 4) * ld + (c & 15) * 4);
  }
#pragma unroll
  for (int i = 0; i < ITERS; ++i) {
    const int c = tid + i * NTHREADS;
    *(v4h*)(dst + (c >> 4) * KC + (c & 15) * 4) =
        (v4h){(_Float16)v[i].x, (_Float16)v[i].y, (_Float16)v[i].z, (_Float16)v[i].w};
  }
}

__device__ __forceinline__ void prefetch_rows_f32(const float* src, int ld, int rows, int tid) {
  if (tid < rows) __builtin_prefetch(src + (size_t)tid * ld, 0, 0);
}

// C/D layout: col = lane&15, row = r0 + (lane>>4)*8 + g.
__device__ __forceinline__ void store_tile_f16(_Float16* out, size_t ld, int r0, int c0,
                                               v8f acc[2][2], int wm, int wn, int lane) {
  const int nn = lane & 15;
  const int rb = r0 + wm * 32 + ((lane >> 4) << 3);
#pragma unroll
  for (int mi = 0; mi < 2; ++mi)
#pragma unroll
    for (int ni = 0; ni < 2; ++ni) {
      const int col = c0 + wn * 32 + ni * 16 + nn;
#pragma unroll
      for (int g = 0; g < 8; ++g)
        out[(size_t)(rb + mi * 16 + g) * ld + col] = (_Float16)acc[mi][ni][g];
    }
}

__device__ __forceinline__ void store_tile_f32(float* out, size_t ld, int r0, int c0,
                                               v8f acc[2][2], int wm, int wn, int lane) {
  const int nn = lane & 15;
  const int rb = r0 + wm * 32 + ((lane >> 4) << 3);
#pragma unroll
  for (int mi = 0; mi < 2; ++mi)
#pragma unroll
    for (int ni = 0; ni < 2; ++ni) {
      const int col = c0 + wn * 32 + ni * 16 + nn;
#pragma unroll
      for (int g = 0; g < 8; ++g)
        out[(size_t)(rb + mi * 16 + g) * ld + col] = acc[mi][ni][g];
    }
}

// ---------------------------------------------------------------------------
// Kernel 1: Q = X @ W_in^T (f32 in). B-panel[e][k] = W_in[e][k] (row-major as
// is). Epilogue also writes Q^T for the mix GEMM.
// ---------------------------------------------------------------------------
__global__ __launch_bounds__(NTHREADS) void k_qproj(const float* __restrict__ X,
                                                    const float* __restrict__ Win,
                                                    _Float16* __restrict__ Qh,
                                                    _Float16* __restrict__ Qth,
                                                    int Nn, int Dd) {
  __shared__ _Float16 As[BM * KC];
  __shared__ _Float16 Bs[BN * KC];
  const int b = blockIdx.z, n0 = blockIdx.x * BM, e0 = blockIdx.y * BN;
  const int tid = threadIdx.x, lane = tid & 31, w = tid >> 5;
  const int wm = w >> 2, wn = w & 3;
  const float* Xb = X + (size_t)b * Nn * Dd;
  v8f acc[2][2] = {};
  for (int k0 = 0; k0 < Dd; k0 += KC) {
    stage_f32<BM>(As, Xb + (size_t)n0 * Dd + k0, Dd, tid);
    stage_f32<BN>(Bs, Win + (size_t)e0 * Dd + k0, Dd, tid);
    if (k0 + KC < Dd) {
      prefetch_rows_f32(Xb + (size_t)n0 * Dd + k0 + KC, Dd, BM, tid);
      prefetch_rows_f32(Win + (size_t)e0 * Dd + k0 + KC, Dd, BN, tid);
    }
    __syncthreads();
    mma_block(As, Bs, acc, wm, wn, lane);
    __syncthreads();
  }
  store_tile_f16(Qh + (size_t)b * Nn * Dd, Dd, n0, e0, acc, wm, wn, lane);
  // transposed copy Q^T[e][n]
  _Float16* Qt = Qth + (size_t)b * Nn * Dd;
  const int nn = lane & 15;
  const int rb = n0 + wm * 32 + ((lane >> 4) << 3);
#pragma unroll
  for (int mi = 0; mi < 2; ++mi)
#pragma unroll
    for (int ni = 0; ni < 2; ++ni) {
      const int col = e0 + wn * 32 + ni * 16 + nn;
#pragma unroll
      for (int g = 0; g < 8; ++g)
        Qt[(size_t)col * Nn + (rb + mi * 16 + g)] = (_Float16)acc[mi][ni][g];
    }
}

// ---------------------------------------------------------------------------
// Kernel 2: S = Q @ Q^T. Both panels are f16 row-major reads of Qh -> TDM-fed
// and double-buffered. Fully-masked tiles (m0 >= n0+64) skipped.
// ---------------------------------------------------------------------------
__global__ __launch_bounds__(NTHREADS) void k_scores(const _Float16* __restrict__ Qh,
                                                     _Float16* __restrict__ S,
                                                     int Nn, int Dd) {
  const int b = blockIdx.z, n0 = blockIdx.x * BM, m0 = blockIdx.y * BN;
  if (m0 >= n0 + BM) return;   // causal: entire tile masked, never read later
  __shared__ _Float16 As[2][BM * KC];
  __shared__ _Float16 Bs[2][BN * KC];
  const int tid = threadIdx.x, lane = tid & 31, w = tid >> 5;
  const int wm = w >> 2, wn = w & 3;
  const _Float16* srcA = Qh + (size_t)b * Nn * Dd + (size_t)n0 * Dd;
  const _Float16* srcB = Qh + (size_t)b * Nn * Dd + (size_t)m0 * Dd;
  v8f acc[2][2] = {};
  const int nk = Dd / KC;
  if (w == 0) {
    tdm_load_tile(lds_off(As[0]), srcA, Dd, BM);
    tdm_load_tile(lds_off(Bs[0]), srcB, Dd, BN);
    __builtin_amdgcn_s_wait_tensorcnt(0);
  }
  __syncthreads();
  for (int i = 0; i < nk; ++i) {
    const int cur = i & 1, nxt = cur ^ 1;
    if (w == 0 && i + 1 < nk) {        // DMA next chunk while computing this one
      tdm_load_tile(lds_off(As[nxt]), srcA + (i + 1) * KC, Dd, BM);
      tdm_load_tile(lds_off(Bs[nxt]), srcB + (i + 1) * KC, Dd, BN);
    }
    mma_block(As[cur], Bs[cur], acc, wm, wn, lane);
    if (w == 0) __builtin_amdgcn_s_wait_tensorcnt(0);
    __syncthreads();
  }
  store_tile_f16(S + (size_t)b * Nn * Nn, Nn, n0, m0, acc, wm, wn, lane);
}

// ---------------------------------------------------------------------------
// Kernel 3: causal softmax in place. Row n attends to m in [0, n); row 0 -> 0.
// ---------------------------------------------------------------------------
__global__ __launch_bounds__(NTHREADS) void k_softmax(_Float16* __restrict__ S, int Nn) {
  const int n = blockIdx.x, b = blockIdx.y;
  _Float16* row = S + ((size_t)b * Nn + n) * Nn;
  const int tid = threadIdx.x;
  if (n == 0) {
    for (int m = tid; m < Nn; m += NTHREADS) row[m] = (_Float16)0.f;
    return;
  }
  __shared__ float red[NTHREADS / 32];
  const int lane = tid & 31, w = tid >> 5;
  float lmax = -1e38f;
  for (int m = tid; m < n; m += NTHREADS) lmax = fmaxf(lmax, (float)row[m]);
#pragma unroll
  for (int o = 16; o > 0; o >>= 1) lmax = fmaxf(lmax, __shfl_xor(lmax, o, 32));
  if (lane == 0) red[w] = lmax;
  __syncthreads();
  float rmax = red[0];
#pragma unroll
  for (int i = 1; i < NTHREADS / 32; ++i) rmax = fmaxf(rmax, red[i]);
  __syncthreads();
  float lsum = 0.f;
  for (int m = tid; m < n; m += NTHREADS) lsum += __expf((float)row[m] - rmax);
#pragma unroll
  for (int o = 16; o > 0; o >>= 1) lsum += __shfl_xor(lsum, o, 32);
  if (lane == 0) red[w] = lsum;
  __syncthreads();
  float rsum = 0.f;
#pragma unroll
  for (int i = 0; i < NTHREADS / 32; ++i) rsum += red[i];
  const float inv = 1.f / rsum;
  for (int m = tid; m < Nn; m += NTHREADS)
    row[m] = (m < n) ? (_Float16)(__expf((float)row[m] - rmax) * inv) : (_Float16)0.f;
}

// ---------------------------------------------------------------------------
// Kernel 4: mix = P @ Q. A = P rows, B-panel[d][m] = Q^T rows; both f16 ->
// TDM double-buffered. Causal: P[n][m]==0 for m>=n => K loop stops at n0+64.
// ---------------------------------------------------------------------------
__global__ __launch_bounds__(NTHREADS) void k_mix(const _Float16* __restrict__ P,
                                                  const _Float16* __restrict__ Qth,
                                                  _Float16* __restrict__ Mixh,
                                                  int Nn, int Dd) {
  __shared__ _Float16 As[2][BM * KC];
  __shared__ _Float16 Bs[2][BN * KC];
  const int b = blockIdx.z, n0 = blockIdx.x * BM, d0 = blockIdx.y * BN;
  const int tid = threadIdx.x, lane = tid & 31, w = tid >> 5;
  const int wm = w >> 2, wn = w & 3;
  const _Float16* srcA = P + (size_t)b * Nn * Nn + (size_t)n0 * Nn;
  const _Float16* srcB = Qth + (size_t)b * Nn * Dd + (size_t)d0 * Nn;
  v8f acc[2][2] = {};
  const int nk = (n0 + BM) / KC;       // strict-lower causal band
  if (w == 0) {
    tdm_load_tile(lds_off(As[0]), srcA, Nn, BM);
    tdm_load_tile(lds_off(Bs[0]), srcB, Nn, BN);
    __builtin_amdgcn_s_wait_tensorcnt(0);
  }
  __syncthreads();
  for (int i = 0; i < nk; ++i) {
    const int cur = i & 1, nxt = cur ^ 1;
    if (w == 0 && i + 1 < nk) {
      tdm_load_tile(lds_off(As[nxt]), srcA + (i + 1) * KC, Nn, BM);
      tdm_load_tile(lds_off(Bs[nxt]), srcB + (i + 1) * KC, Nn, BN);
    }
    mma_block(As[cur], Bs[cur], acc, wm, wn, lane);
    if (w == 0) __builtin_amdgcn_s_wait_tensorcnt(0);
    __syncthreads();
  }
  store_tile_f16(Mixh + (size_t)b * Nn * Dd, Dd, n0, d0, acc, wm, wn, lane);
}

// ---------------------------------------------------------------------------
// Kernel 5: out = [mix, q] @ W_out^T. A-source switches between mix (k < D)
// and q (k >= D) panels, TDM-fed; B-panel (W_out, f32) register-staged.
// ---------------------------------------------------------------------------
__global__ __launch_bounds__(NTHREADS) void k_out(const _Float16* __restrict__ Mixh,
                                                  const _Float16* __restrict__ Qh,
                                                  const float* __restrict__ Wout,
                                                  float* __restrict__ Out,
                                                  int Nn, int Dd) {
  __shared__ _Float16 As[BM * KC];
  __shared__ _Float16 Bs[BN * KC];
  const int b = blockIdx.z, n0 = blockIdx.x * BM, d0 = blockIdx.y * BN;
  const int tid = threadIdx.x, lane = tid & 31, w = tid >> 5;
  const int wm = w >> 2, wn = w & 3;
  const size_t baseA = (size_t)b * Nn * Dd + (size_t)n0 * Dd;
  const int K2 = 2 * Dd;
  v8f acc[2][2] = {};
  for (int k0 = 0; k0 < K2; k0 += KC) {
    const _Float16* Asrc = (k0 < Dd) ? (Mixh + baseA + k0) : (Qh + baseA + (k0 - Dd));
    if (w == 0) tdm_load_tile(lds_off(As), Asrc, Dd, BM);
    stage_f32<BN>(Bs, Wout + (size_t)d0 * K2 + k0, K2, tid);
    if (k0 + KC < K2)
      prefetch_rows_f32(Wout + (size_t)d0 * K2 + k0 + KC, K2, BN, tid);
    if (w == 0) __builtin_amdgcn_s_wait_tensorcnt(0);
    __syncthreads();
    mma_block(As, Bs, acc, wm, wn, lane);
    __syncthreads();
  }
  store_tile_f32(Out + (size_t)b * Nn * Dd, Dd, n0, d0, acc, wm, wn, lane);
}

// ---------------------------------------------------------------------------
extern "C" void kernel_launch(void* const* d_in, const int* in_sizes, int n_in,
                              void* d_out, int out_size, void* d_ws, size_t ws_size,
                              hipStream_t stream) {
  (void)in_sizes; (void)n_in; (void)out_size; (void)ws_size;
  const int B = 16, N = 2048, D = 512;

  const float* query = (const float*)d_in[0];   // [B,N,D] f32
  const float* Win   = (const float*)d_in[1];   // [D,D]   f32
  const float* Wout  = (const float*)d_in[2];   // [D,2D]  f32
  float* out = (float*)d_out;                   // [B,N,D] f32

  _Float16* Qh   = (_Float16*)d_ws;                 // B*N*D halves
  _Float16* Qth  = Qh  + (size_t)B * N * D;         // B*D*N halves
  _Float16* Mixh = Qth + (size_t)B * N * D;         // B*N*D halves
  _Float16* S    = Mixh + (size_t)B * N * D;        // B*N*N halves

  const dim3 blk(NTHREADS);
  k_qproj  <<<dim3(N / BM, D / BN, B), blk, 0, stream>>>(query, Win, Qh, Qth, N, D);
  k_scores <<<dim3(N / BM, N / BN, B), blk, 0, stream>>>(Qh, S, N, D);
  k_softmax<<<dim3(N, B),              blk, 0, stream>>>(S, N);
  k_mix    <<<dim3(N / BM, D / BN, B), blk, 0, stream>>>(S, Qth, Mixh, N, D);
  k_out    <<<dim3(N / BM, D / BN, B), blk, 0, stream>>>(Mixh, Qh, Wout, out, N, D);
}
